// SelfAttentionBlock_12807592477313
// MI455X (gfx1250) — matrix operand order repaired and checked by reading, not compiled, not run on hardware
//
#include <hip/hip_runtime.h>
#include <hip/hip_bf16.h>
#include <cstdint>
#include <cstddef>

#define HIDDEN   1024
#define HEADS    16
#define HEAD_DIM 64
#define BATCH    4
#define SEQ      2048
#define ROWS     (BATCH * SEQ)   // 8192
#define LN_EPS   1e-5f

// ---------------------------------------------------------------------------
// WMMA fragment types (CDNA5 wave32, 16x16x32 bf16 -> f32)
// ---------------------------------------------------------------------------
typedef __attribute__((ext_vector_type(16))) __bf16 bf16x16;
typedef __attribute__((ext_vector_type(8)))  float  f32x8;
typedef __attribute__((ext_vector_type(4)))  int    i32x4;

union FragAB { unsigned u[8]; bf16x16 v; };
union FragC  { float    f[8]; f32x8   v; };

static __device__ __forceinline__ unsigned short f2bf(float f) {
  union { float f; unsigned u; } un; un.f = f;
  unsigned r = un.u + 0x7FFFu + ((un.u >> 16) & 1u);   // round-to-nearest-even
  return (unsigned short)(r >> 16);
}

static __device__ __forceinline__ void wmma_bf16(FragC& c, const FragAB& a, const FragAB& b) {
  // 8 args: (neg_a, A, neg_b, B, c_mod, C, reuse_a, reuse_b)
  c.v = __builtin_amdgcn_wmma_f32_16x16x32_bf16(false, a.v, false, b.v, (short)0, c.v, false, false);
}

// ---------------------------------------------------------------------------
// Async global->LDS copy (CDNA5 GLOBAL_LOAD_ASYNC_TO_LDS_B128, ASYNCcnt)
// Builtin signature (from probe diagnostic): (AS1 i32x4*, AS3 i32x4*, imm, imm)
// ---------------------------------------------------------------------------
#if __has_builtin(__builtin_amdgcn_global_load_async_to_lds_b128)
#define ASYNC_LDS 1
typedef __attribute__((address_space(1))) i32x4 gi32x4;
typedef __attribute__((address_space(3))) i32x4 li32x4;
#else
#define ASYNC_LDS 0
#endif

static __device__ __forceinline__ void async_copy_b128(const unsigned short* g, unsigned short* l) {
#if ASYNC_LDS
  __builtin_amdgcn_global_load_async_to_lds_b128((gi32x4*)g, (li32x4*)l, 0, 0);
#else
  *(uint4*)l = *(const uint4*)g;
#endif
}

static __device__ __forceinline__ void async_wait() {
#if ASYNC_LDS
#if __has_builtin(__builtin_amdgcn_s_wait_asynccnt)
  __builtin_amdgcn_s_wait_asynccnt(0);
#else
  asm volatile("s_wait_asynccnt 0" ::: "memory");
#endif
#endif
}

// ---------------------------------------------------------------------------
// Fragment loaders (m/n layout per CDNA5 ISA §7.12.2; this code's k order is contiguous, k = 16h + s)
// ---------------------------------------------------------------------------
// A fragment: 16(M) x 32(K), src row-major [m][k] bf16 with leading dim `ld`.
static __device__ __forceinline__ void load_a_frag(const unsigned short* src, int ld, FragAB& fr) {
  int lane = threadIdx.x & 31;
  int m  = lane & 15;
  int kh = (lane >> 4) << 4;          // 0 or 16
  const unsigned short* row = src + m * ld;
#pragma unroll
  for (int v = 0; v < 8; ++v) {
    int kk = kh + (v << 1);   // K pair index
    fr.u[v] = *(const unsigned*)(row + kk);
  }
}

// B fragment: 32(K) x 16(N); storage K-contiguous: element (k,n) at src[n*ld + k].
static __device__ __forceinline__ void load_b_frag(const unsigned short* src, int ld, FragAB& fr) {
  int lane = threadIdx.x & 31;
  int n  = lane & 15;
  int kh = (lane >> 4) << 4;          // 0 or 16
  const unsigned short* col = src + n * ld + kh;
#pragma unroll
  for (int v = 0; v < 8; ++v) {
    fr.u[v] = *(const unsigned*)(col + (v << 1));
  }
}

// B fragment from natural row-major [K][N] storage: element (k,n) at src[k*ld + n].
static __device__ __forceinline__ void load_b_frag_strided(const unsigned short* src, int ld, FragAB& fr) {
  int lane = threadIdx.x & 31;
  int n  = lane & 15;
  int kh = (lane >> 4) << 4;          // 0 or 16
#pragma unroll
  for (int v = 0; v < 8; ++v) {
    int k = kh + (v << 1);
    unsigned lo = src[k * ld + n];
    unsigned hi = src[(k + 1) * ld + n];
    fr.u[v] = lo | (hi << 16);
  }
}

// ---------------------------------------------------------------------------
// 1) LayerNorm: one block (128 threads) per row of 1024 floats
// ---------------------------------------------------------------------------
__global__ __launch_bounds__(128) void ln_kernel(const float* __restrict__ x,
                                                 const float* __restrict__ gamma,
                                                 const float* __restrict__ beta,
                                                 float* __restrict__ xn,
                                                 unsigned short* __restrict__ xnb) {
  __shared__ float rs[4], rss[4];
  int row = blockIdx.x;
  int tid = threadIdx.x;
  const float4* xr = (const float4*)(x + (size_t)row * HIDDEN);
  float4 v0 = xr[tid];
  float4 v1 = xr[tid + 128];
  float s  = v0.x + v0.y + v0.z + v0.w + v1.x + v1.y + v1.z + v1.w;
  float ss = v0.x*v0.x + v0.y*v0.y + v0.z*v0.z + v0.w*v0.w
           + v1.x*v1.x + v1.y*v1.y + v1.z*v1.z + v1.w*v1.w;
#pragma unroll
  for (int m = 1; m < 32; m <<= 1) { s += __shfl_xor(s, m, 32); ss += __shfl_xor(ss, m, 32); }
  int wave = tid >> 5, lane = tid & 31;
  if (lane == 0) { rs[wave] = s; rss[wave] = ss; }
  __syncthreads();
  s  = rs[0] + rs[1] + rs[2] + rs[3];
  ss = rss[0] + rss[1] + rss[2] + rss[3];
  float mean = s * (1.0f / HIDDEN);
  float var  = ss * (1.0f / HIDDEN) - mean * mean;
  float inv  = rsqrtf(var + LN_EPS);
  float* xo = xn + (size_t)row * HIDDEN;
  unsigned short* xb = xnb + (size_t)row * HIDDEN;
  float vals[8] = {v0.x, v0.y, v0.z, v0.w, v1.x, v1.y, v1.z, v1.w};
#pragma unroll
  for (int e = 0; e < 8; ++e) {
    int col = (e < 4) ? (tid * 4 + e) : ((tid + 128) * 4 + (e - 4));
    float y = (vals[e] - mean) * inv * gamma[col] + beta[col];
    xo[col] = y;
    xb[col] = f2bf(y);
  }
}

// ---------------------------------------------------------------------------
// 2) Weight convert + transpose: W[k][n] f32 -> Wt[n][k] bf16 (K-contiguous)
// ---------------------------------------------------------------------------
__global__ __launch_bounds__(256) void wcvt_kernel(const float* __restrict__ W,
                                                   unsigned short* __restrict__ Wt) {
  int idx = blockIdx.x * 256 + threadIdx.x;        // < 1<<20
  int k = idx >> 10, n = idx & 1023;
  Wt[(size_t)n * HIDDEN + k] = f2bf(W[idx]);
}

// ---------------------------------------------------------------------------
// 3) QKV GEMM: 128x64 block tile, K-stage 64, 4 waves each with 2x4 WMMA accs
//    -> 16 v_wmma per barrier pair; async global->LDS staging
// ---------------------------------------------------------------------------
#define GK   64
#define GLDA 72   // 64 + 8 pad -> 144B rows (16B aligned, bank-spread)
__global__ __launch_bounds__(128) void qkv_gemm_kernel(const unsigned short* __restrict__ A,
                                                       const unsigned short* __restrict__ Wt,
                                                       const float* __restrict__ bias,
                                                       unsigned short* __restrict__ dst) {
  __shared__ unsigned short sA[128][GLDA];   // [m][k]
  __shared__ unsigned short sB[64][GLDA];    // [n][k] (transposed in global)
  int tid  = threadIdx.x;
  int wave = tid >> 5;
  int m0 = blockIdx.y << 7;
  int n0 = blockIdx.x << 6;
  int mw = wave << 5;

  FragC c[2][4];
#pragma unroll
  for (int i = 0; i < 2; ++i)
#pragma unroll
    for (int j = 0; j < 4; ++j)
#pragma unroll
      for (int r = 0; r < 8; ++r) c[i][j].f[r] = 0.0f;

  for (int k0 = 0; k0 < HIDDEN; k0 += GK) {
#pragma unroll
    for (int it = 0; it < 8; ++it) {                 // A: 128 rows x 8 chunks
      int c8 = tid + (it << 7);
      int r = c8 >> 3, cc = (c8 & 7) << 3;
      async_copy_b128(A + (size_t)(m0 + r) * HIDDEN + k0 + cc, &sA[r][cc]);
    }
#pragma unroll
    for (int it = 0; it < 4; ++it) {                 // B: 64 rows x 8 chunks
      int c8 = tid + (it << 7);
      int r = c8 >> 3, cc = (c8 & 7) << 3;
      async_copy_b128(Wt + (size_t)(n0 + r) * HIDDEN + k0 + cc, &sB[r][cc]);
    }
    if (k0 + GK < HIDDEN) {                          // hint next stage into L2
      __builtin_prefetch(A  + (size_t)(m0 + (tid >> 3)) * HIDDEN + k0 + GK, 0, 1);
      __builtin_prefetch(Wt + (size_t)(n0 + (tid >> 3)) * HIDDEN + k0 + GK, 0, 1);
    }
    async_wait();
    __syncthreads();

#pragma unroll
    for (int ks = 0; ks < GK; ks += 32) {
      FragAB a[2], b[4];
      load_a_frag(&sA[mw][ks],      GLDA, a[0]);
      load_a_frag(&sA[mw + 16][ks], GLDA, a[1]);
#pragma unroll
      for (int j = 0; j < 4; ++j) load_b_frag(&sB[j << 4][ks], GLDA, b[j]);
#pragma unroll
      for (int i = 0; i < 2; ++i)
#pragma unroll
        for (int j = 0; j < 4; ++j) wmma_bf16(c[i][j], a[i], b[j]);
    }
    __syncthreads();
  }

  int lane = tid & 31;
  int hh = lane >> 4;        // lane half -> M += 8
  int nl = lane & 15;
#pragma unroll
  for (int i = 0; i < 2; ++i)
#pragma unroll
    for (int j = 0; j < 4; ++j) {
      int n = n0 + (j << 4) + nl;
      float bv = bias[n];
      int head = n >> 6, d = n & 63;
#pragma unroll
      for (int r = 0; r < 8; ++r) {
        int m = m0 + mw + (i << 4) + r + (hh << 3);
        int b = m >> 11, sq = m & 2047;
        dst[((size_t)((b * HEADS + head) * SEQ + sq) << 6) + d] = f2bf(c[i][j].f[r] + bv);
      }
    }
}

// ---------------------------------------------------------------------------
// 4) Flash attention + residual: block = (bh, 64-query tile), 4 waves x 16 rows
// ---------------------------------------------------------------------------
#define ALD 72   // 64 + 8 pad -> 144B rows
__global__ __launch_bounds__(128) void attn_kernel(const unsigned short* __restrict__ Q,
                                                   const unsigned short* __restrict__ K,
                                                   const unsigned short* __restrict__ V,
                                                   const float* __restrict__ xn,
                                                   float* __restrict__ out) {
  __shared__ unsigned short sK[64][ALD];       // [key][dim]
  __shared__ unsigned short sV[64][ALD];       // [key][dim]
  __shared__ unsigned short sP[4][16][ALD];    // per-wave P tile, [row][key]
  int tid  = threadIdx.x;
  int wave = tid >> 5;
  int lane = tid & 31;
  int hh = lane >> 4;
  int nl = lane & 15;
  int q0 = blockIdx.x << 6;
  int bh = blockIdx.y;
  size_t basebh = (size_t)bh * SEQ * HEAD_DIM;

  // persistent Q fragments (16 rows x dims 0..31 / 32..63)
  FragAB qa[2];
  load_a_frag(Q + basebh + (size_t)(q0 + (wave << 4)) * HEAD_DIM,      HEAD_DIM, qa[0]);
  load_a_frag(Q + basebh + (size_t)(q0 + (wave << 4)) * HEAD_DIM + 32, HEAD_DIM, qa[1]);

  float mi[8], li[8];
  FragC o[4];
#pragma unroll
  for (int r = 0; r < 8; ++r) { mi[r] = -1e30f; li[r] = 0.0f; }
#pragma unroll
  for (int t = 0; t < 4; ++t)
#pragma unroll
    for (int r = 0; r < 8; ++r) o[t].f[r] = 0.0f;

  for (int kt = 0; kt < SEQ / 64; ++kt) {
    const unsigned short* kg = K + basebh + (size_t)(kt << 6) * HEAD_DIM;
    const unsigned short* vg = V + basebh + (size_t)(kt << 6) * HEAD_DIM;
#pragma unroll
    for (int it = 0; it < 4; ++it) {                 // 64 rows x 8 chunks of 8 bf16
      int c8 = tid + (it << 7);
      int r = c8 >> 3, cc = (c8 & 7) << 3;
      async_copy_b128(kg + (size_t)r * HEAD_DIM + cc, &sK[r][cc]);
      async_copy_b128(vg + (size_t)r * HEAD_DIM + cc, &sV[r][cc]);
    }
    async_wait();
    __syncthreads();

    // scores: Q(16x64) @ K^T -> 4 x 16x16 f32 tiles per wave
    FragC sc[4];
#pragma unroll
    for (int t = 0; t < 4; ++t) {
#pragma unroll
      for (int r = 0; r < 8; ++r) sc[t].f[r] = 0.0f;
      FragAB b0, b1;
      load_b_frag(&sK[t << 4][0],  ALD, b0);   // keys t*16.. x dims 0..31
      load_b_frag(&sK[t << 4][32], ALD, b1);   // dims 32..63
      wmma_bf16(sc[t], qa[0], b0);
      wmma_bf16(sc[t], qa[1], b1);
#pragma unroll
      for (int r = 0; r < 8; ++r) sc[t].f[r] *= 0.125f;   // 1/sqrt(64)
    }

    // online softmax: C-layout row m = r + 8*hh lives across lanes of one half
#pragma unroll
    for (int r = 0; r < 8; ++r) {
      float mx = fmaxf(fmaxf(sc[0].f[r], sc[1].f[r]), fmaxf(sc[2].f[r], sc[3].f[r]));
#pragma unroll
      for (int msk = 1; msk < 16; msk <<= 1) mx = fmaxf(mx, __shfl_xor(mx, msk, 32));
      float mnew = fmaxf(mi[r], mx);
      float corr = __expf(mi[r] - mnew);
      mi[r] = mnew;
      float psum = 0.0f;
#pragma unroll
      for (int t = 0; t < 4; ++t) {
        float p = __expf(sc[t].f[r] - mnew);
        psum += p;
        sP[wave][r + (hh << 3)][(t << 4) + nl] = f2bf(p);
      }
#pragma unroll
      for (int msk = 1; msk < 16; msk <<= 1) psum += __shfl_xor(psum, msk, 32);
      li[r] = li[r] * corr + psum;
#pragma unroll
      for (int t = 0; t < 4; ++t) o[t].f[r] *= corr;
    }
    __syncthreads();

    // o += P(16x64) @ V(64x64)
    FragAB pa[2];
    load_a_frag(&sP[wave][0][0],  ALD, pa[0]);
    load_a_frag(&sP[wave][0][32], ALD, pa[1]);
#pragma unroll
    for (int t = 0; t < 4; ++t) {
      FragAB b0, b1;
      load_b_frag_strided(&sV[0][t << 4],  ALD, b0);   // keys 0..31  x dims t*16..
      load_b_frag_strided(&sV[32][t << 4], ALD, b1);   // keys 32..63
      wmma_bf16(o[t], pa[0], b0);
      wmma_bf16(o[t], pa[1], b1);
    }
    __syncthreads();
  }

  // epilogue: normalize, add residual, write f32 output
  int b = bh >> 4, h = bh & 15;
#pragma unroll
  for (int t = 0; t < 4; ++t) {
    int col = (h << 6) + (t << 4) + nl;
#pragma unroll
    for (int r = 0; r < 8; ++r) {
      int srow = q0 + (wave << 4) + r + (hh << 3);
      size_t gi = ((size_t)(b * SEQ + srow) << 10) + col;
      out[gi] = xn[gi] + o[t].f[r] / li[r];
    }
  }
}

// ---------------------------------------------------------------------------
// Launch
// ---------------------------------------------------------------------------
extern "C" void kernel_launch(void* const* d_in, const int* in_sizes, int n_in,
                              void* d_out, int out_size, void* d_ws, size_t ws_size,
                              hipStream_t stream) {
  (void)in_sizes; (void)n_in; (void)out_size; (void)ws_size;
  const float* x     = (const float*)d_in[0];
  const float* Wq    = (const float*)d_in[1];
  const float* bq    = (const float*)d_in[2];
  const float* Wk    = (const float*)d_in[3];
  const float* bk    = (const float*)d_in[4];
  const float* Wv    = (const float*)d_in[5];
  const float* bv    = (const float*)d_in[6];
  const float* gamma = (const float*)d_in[7];
  const float* beta  = (const float*)d_in[8];
  float* out = (float*)d_out;

  // workspace carve-up (~102 MB total)
  char* ws = (char*)d_ws;
  float*          xn_f32 = (float*)ws;          ws += (size_t)ROWS * HIDDEN * 4;
  unsigned short* xn_bf  = (unsigned short*)ws; ws += (size_t)ROWS * HIDDEN * 2;
  unsigned short* Wtq    = (unsigned short*)ws; ws += (size_t)HIDDEN * HIDDEN * 2;
  unsigned short* Wtk    = (unsigned short*)ws; ws += (size_t)HIDDEN * HIDDEN * 2;
  unsigned short* Wtv    = (unsigned short*)ws; ws += (size_t)HIDDEN * HIDDEN * 2;
  unsigned short* qb     = (unsigned short*)ws; ws += (size_t)ROWS * HIDDEN * 2;
  unsigned short* kb     = (unsigned short*)ws; ws += (size_t)ROWS * HIDDEN * 2;
  unsigned short* vb     = (unsigned short*)ws; ws += (size_t)ROWS * HIDDEN * 2;

  ln_kernel<<<ROWS, 128, 0, stream>>>(x, gamma, beta, xn_f32, xn_bf);

  int wblocks = (HIDDEN * HIDDEN) / 256;
  wcvt_kernel<<<wblocks, 256, 0, stream>>>(Wq, Wtq);
  wcvt_kernel<<<wblocks, 256, 0, stream>>>(Wk, Wtk);
  wcvt_kernel<<<wblocks, 256, 0, stream>>>(Wv, Wtv);

  dim3 gg(HIDDEN / 64, ROWS / 128);
  qkv_gemm_kernel<<<gg, 128, 0, stream>>>(xn_bf, Wtq, bq, qb);
  qkv_gemm_kernel<<<gg, 128, 0, stream>>>(xn_bf, Wtk, bk, kb);
  qkv_gemm_kernel<<<gg, 128, 0, stream>>>(xn_bf, Wtv, bv, vb);

  attn_kernel<<<dim3(SEQ / 64, BATCH * HEADS), 128, 0, stream>>>(qb, kb, vb, xn_f32, out);
}
